// CausalSelfAttention_89833535964077
// MI455X (gfx1250) — compile-verified, matrix operand order checked by reading, not run on hardware
//
#include <hip/hip_runtime.h>
#include <hip/hip_bf16.h>

// ---------------------------------------------------------------------------
// CDNA5 (gfx1250) fused causal self-attention, bf16 WMMA pipeline with
// async-to-LDS double-buffered staging.
//   K0: convert x / w_qkv / w_proj fp32 -> bf16 (one pass, halves L2 tile BW)
//   K1: qkv = x @ w_qkv^T  (bf16 WMMA, async staging) -> Q/K/V [B,H,T,64]
//   K2: flash attention (online softmax, causal); K async-staged row-major,
//       V register-prefetched and stored transposed in LDS so every WMMA
//       fragment load is a contiguous b128.
//   K3: out = attn @ w_proj^T (bf16 WMMA, async staging) -> fp32
// ---------------------------------------------------------------------------

typedef __bf16 bf16_t;
typedef __bf16 v16bf __attribute__((ext_vector_type(16)));
typedef __bf16 v8bf  __attribute__((ext_vector_type(8)));
typedef float  v8f   __attribute__((ext_vector_type(8)));

#define WMMA_BF16(a, b, c) \
  __builtin_amdgcn_wmma_f32_16x16x32_bf16(false, (a), false, (b), (short)0, (c), false, false)

// Wait until at most N async(-to-LDS) ops from this wave remain outstanding.
#define ASYNC_WAIT(nstr) asm volatile("s_wait_asynccnt " nstr ::: "memory")
// Wave-local LDS RAW fence (DS ops are in-order per wave).
#define DS_FENCE() asm volatile("s_wait_dscnt 0" ::: "memory")

constexpr long QKV_ELEMS = (long)4 * 16 * 2048 * 64;  // 8388608 per Q/K/V

// One wave-wide 16-byte async copy: global -> LDS, tracked by ASYNCcnt.
__device__ __forceinline__ void async_b128(const bf16_t* gsrc, const bf16_t* lds_dst) {
  const unsigned int       laddr = (unsigned int)(uintptr_t)lds_dst;   // low 32 = LDS offset
  const unsigned long long gaddr = (unsigned long long)(uintptr_t)gsrc;
  asm volatile("global_load_async_to_lds_b128 %0, %1, off"
               :: "v"(laddr), "v"(gaddr) : "memory");
}

// ---------------------------------------------------------------------------
// Kernel 0: fp32 -> bf16 conversion (8 elements per thread, b128 stores).
// ---------------------------------------------------------------------------
__global__ __launch_bounds__(256)
void cvt_bf16_kernel(const float* __restrict__ src, bf16_t* __restrict__ dst) {
  const long i = (long)blockIdx.x * 256 + threadIdx.x;
  const float4 a = ((const float4*)src)[2 * i];
  const float4 b = ((const float4*)src)[2 * i + 1];
  v8bf o;
  o[0] = (bf16_t)a.x; o[1] = (bf16_t)a.y; o[2] = (bf16_t)a.z; o[3] = (bf16_t)a.w;
  o[4] = (bf16_t)b.x; o[5] = (bf16_t)b.y; o[6] = (bf16_t)b.z; o[7] = (bf16_t)b.w;
  ((v8bf*)dst)[i] = o;
}

// Stage one 128x32 bf16 tile into LDS via async copies (2 b128 per thread).
__device__ __forceinline__ void stage_tile_async(const bf16_t* __restrict__ g,
                                                 long ldg, long row0, int k0,
                                                 bf16_t (*lds)[48], int tid) {
#pragma unroll
  for (int i = 0; i < 2; ++i) {
    const int idx = tid + i * 256;       // 0..511
    const int r = idx >> 2;              // 0..127
    const int c = (idx & 3) * 8;         // 0,8,16,24
    async_b128(g + (row0 + r) * ldg + k0 + c, &lds[r][c]);
  }
}

// Load a 16x32 bf16 WMMA A/B fragment from an LDS tile.
// lane<16: row=l16, K {0..7,16..23}; lane>=16: row=l16, K {8..15,24..31}.
__device__ __forceinline__ v16bf load_frag(const bf16_t (*lds)[48], int row0,
                                           int l16, int half) {
  v16bf f;
  const int r = row0 + l16;
#pragma unroll
  for (int g = 0; g < 2; ++g)
#pragma unroll
    for (int e = 0; e < 8; ++e)
      f[g * 8 + e] = lds[r][half * 8 + g * 16 + e];
  return f;
}

// ---------------------------------------------------------------------------
// Kernel 1: qkv GEMM (bf16 x bf16). A = xb [8192,1024], B = wqb [3072,1024].
// 128x128 block tile, 8 waves in 2(M)x4(N), wave = 64x32 (4x2 WMMA tiles).
// Double-buffered async staging: each wave keeps 4 async ops in flight.
// ---------------------------------------------------------------------------
__global__ __launch_bounds__(256)
void qkv_gemm_kernel(const bf16_t* __restrict__ A, const bf16_t* __restrict__ W,
                     bf16_t* __restrict__ Qb, bf16_t* __restrict__ Kb,
                     bf16_t* __restrict__ Vb)
{
  __shared__ __align__(16) bf16_t As[2][128][48];
  __shared__ __align__(16) bf16_t Bs[2][128][48];

  const int tid  = threadIdx.x;
  const int lane = tid & 31;
  const int w    = tid >> 5;
  const int wm   = w >> 2, wn = w & 3;
  const int half = lane >> 4, l16 = lane & 15;
  const int mb   = blockIdx.x;   // 0..63
  const int nb   = blockIdx.y;   // 0..23

  v8f acc[4][2];
#pragma unroll
  for (int i = 0; i < 4; ++i)
#pragma unroll
    for (int j = 0; j < 2; ++j)
#pragma unroll
      for (int e = 0; e < 8; ++e) acc[i][j][e] = 0.0f;

  stage_tile_async(A, 1024, (long)mb * 128, 0, As[0], tid);
  stage_tile_async(W, 1024, (long)nb * 128, 0, Bs[0], tid);

  for (int kt = 0; kt < 32; ++kt) {           // 32 chunks of K=32
    const int cur = kt & 1;
    if (kt + 1 < 32) {
      stage_tile_async(A, 1024, (long)mb * 128, (kt + 1) * 32, As[1 - cur], tid);
      stage_tile_async(W, 1024, (long)nb * 128, (kt + 1) * 32, Bs[1 - cur], tid);
      ASYNC_WAIT("0x4");                      // current chunk landed; next in flight
    } else {
      ASYNC_WAIT("0x0");
    }
    __syncthreads();

    v16bf afr[4], bfr[2];
#pragma unroll
    for (int i = 0; i < 4; ++i) afr[i] = load_frag(As[cur], wm * 64 + i * 16, l16, half);
#pragma unroll
    for (int j = 0; j < 2; ++j) bfr[j] = load_frag(Bs[cur], wn * 32 + j * 16, l16, half);
#pragma unroll
    for (int i = 0; i < 4; ++i)
#pragma unroll
      for (int j = 0; j < 2; ++j)
        acc[i][j] = WMMA_BF16(afr[i], bfr[j], acc[i][j]);
    __syncthreads();                          // reads done before buffer reuse
  }

  // Scatter epilogue into Q/K/V [B,H,T,64] bf16.
#pragma unroll
  for (int i = 0; i < 4; ++i) {
#pragma unroll
    for (int j = 0; j < 2; ++j) {
      const int n     = nb * 128 + wn * 32 + j * 16 + l16;  // 0..3071
      const int which = n >> 10;
      const int hc    = n & 1023;
      const int h     = hc >> 6;
      const int d     = hc & 63;
      bf16_t* dst = (which == 0) ? Qb : ((which == 1) ? Kb : Vb);
#pragma unroll
      for (int v = 0; v < 8; ++v) {
        const int m = mb * 128 + wm * 64 + i * 16 + v + 8 * half;
        const int b = m >> 11, t = m & 2047;
        dst[(((long)(b * 16 + h)) * 2048 + t) * 64 + d] = (bf16_t)acc[i][j][v];
      }
    }
  }
}

// ---------------------------------------------------------------------------
// Kernel 2: causal flash attention.
//   K: async double-buffered, row-major [key][feature] (key-per-lane frags).
//   V: register-prefetched from global (overlaps compute), stored transposed
//      [feature][key] in LDS so P*V B-fragments are contiguous b128 loads.
// One block per (b, h, 128-query tile); each of 8 waves owns 16 query rows.
// ---------------------------------------------------------------------------
__global__ __launch_bounds__(256)
void flash_attn_kernel(const bf16_t* __restrict__ Qb, const bf16_t* __restrict__ Kb,
                       const bf16_t* __restrict__ Vb, bf16_t* __restrict__ Ob)
{
  __shared__ __align__(16) bf16_t Ks[2][64][72];   // [key][feature], 144B rows
  __shared__ __align__(16) bf16_t Vt[2][64][72];   // [feature][key], transposed
  __shared__ bf16_t Ps[8][16][36];                 // per-wave P transpose scratch

  const int tid  = threadIdx.x;
  const int lane = tid & 31;
  const int w    = tid >> 5;
  const int half = lane >> 4, l16 = lane & 15;

  const int bid = blockIdx.x;        // (b*16 + h)*16 + qt
  const int qt  = bid & 15;
  const int bh  = bid >> 4;
  const int b   = bh >> 4;
  const int h   = bh & 15;
  const int q0  = qt * 128 + w * 16;

  const long head_off = ((long)(b * 16 + h)) * 2048 * 64;
  const bf16_t* Qh = Qb + head_off;
  const bf16_t* Kh = Kb + head_off;
  const bf16_t* Vh = Vb + head_off;

  // This thread's two 8-element slices of a 64x64 chunk: (r, c..c+7).
  const int sl_r0 = tid >> 3;               // 0..31
  const int sl_r1 = (tid >> 3) + 32;        // 32..63
  const int sl_c  = (tid & 7) * 8;          // 0..56

  // Async-stage one 64-key K chunk (2 b128 per thread = 2 per wave in flight).
  auto stage_k = [&](int key0, int buf) {
    async_b128(Kh + (long)(key0 + sl_r0) * 64 + sl_c, &Ks[buf][sl_r0][sl_c]);
    async_b128(Kh + (long)(key0 + sl_r1) * 64 + sl_c, &Ks[buf][sl_r1][sl_c]);
  };
  // Prefetch one 64-key V chunk into registers (tracked by LOADcnt).
  auto load_v = [&](int key0, v8bf* vr) {
    vr[0] = *(const v8bf*)(Vh + (long)(key0 + sl_r0) * 64 + sl_c);
    vr[1] = *(const v8bf*)(Vh + (long)(key0 + sl_r1) * 64 + sl_c);
  };

  // Q fragments (rows q0..q0+15, K split 0..31 / 32..63), loaded once.
  v16bf qfr[2];
  {
    const int row = q0 + l16;
#pragma unroll
    for (int c = 0; c < 2; ++c)
#pragma unroll
      for (int g = 0; g < 2; ++g)
#pragma unroll
        for (int e = 0; e < 8; ++e)
          qfr[c][g * 8 + e] = Qh[(long)row * 64 + c * 32 + half * 8 + g * 16 + e];
  }

  float m_row[8], l_row[8];
  v8f o[4];
#pragma unroll
  for (int v = 0; v < 8; ++v) { m_row[v] = -3.0e38f; l_row[v] = 0.0f; }
#pragma unroll
  for (int n = 0; n < 4; ++n)
#pragma unroll
    for (int v = 0; v < 8; ++v) o[n][v] = 0.0f;

  const int   nchunk = 2 * (qt + 1);           // causal bound (uniform per block)
  const float scale  = 0.125f;                 // 1/sqrt(64)
  const float LOG2E  = 1.4426950408889634f;

  v8bf vcur[2], vnxt[2];
  stage_k(0, 0);
  load_v(0, vcur);

  for (int kc = 0; kc < nchunk; ++kc) {
    const int cur  = kc & 1;
    const int key0 = kc * 64;

    // Scatter this chunk's V registers transposed into LDS: Vt[feat][key].
#pragma unroll
    for (int e = 0; e < 8; ++e) {
      Vt[cur][sl_c + e][sl_r0] = vcur[0][e];
      Vt[cur][sl_c + e][sl_r1] = vcur[1][e];
    }
    if (kc + 1 < nchunk) {
      stage_k((kc + 1) * 64, 1 - cur);       // async, overlaps compute
      load_v((kc + 1) * 64, vnxt);           // to regs, overlaps compute
      ASYNC_WAIT("0x2");
    } else {
      ASYNC_WAIT("0x0");
    }
    __syncthreads();                          // also waits our DS stores

    for (int s32 = 0; s32 < 2; ++s32) {
      const int kbase = key0 + s32 * 32;
      if (kbase <= q0 + 15) {                 // wave-local causal skip
        // ---- S = Q K^T : two 16x16 score tiles (32 keys) ----
        v8f st[2];
#pragma unroll
        for (int t16 = 0; t16 < 2; ++t16) {
          v8f sacc;
#pragma unroll
          for (int v = 0; v < 8; ++v) sacc[v] = 0.0f;
          const int kl = s32 * 32 + t16 * 16 + l16;   // local key (= B col)
#pragma unroll
          for (int c = 0; c < 2; ++c) {
            v16bf kb;
#pragma unroll
            for (int g = 0; g < 2; ++g)
#pragma unroll
              for (int e = 0; e < 8; ++e)
                kb[g * 8 + e] = Ks[cur][kl][c * 32 + half * 8 + g * 16 + e];
            sacc = WMMA_BF16(qfr[c], kb, sacc);
          }
          st[t16] = sacc;
        }

        // ---- online softmax over 32 keys ----
#pragma unroll
        for (int v = 0; v < 8; ++v) {
          const int row = q0 + v + 8 * half;
          float s0 = st[0][v] * scale;
          float s1 = st[1][v] * scale;
          if (kbase + l16 > row)      s0 = -3.0e38f;
          if (kbase + 16 + l16 > row) s1 = -3.0e38f;
          float mx = fmaxf(s0, s1);
#pragma unroll
          for (int off = 1; off < 16; off <<= 1)
            mx = fmaxf(mx, __shfl_xor(mx, off, 32));   // stays within half
          const float nm    = fmaxf(m_row[v], mx);
          const float alpha = exp2f((m_row[v] - nm) * LOG2E);
          const float p0    = exp2f((s0 - nm) * LOG2E);
          const float p1    = exp2f((s1 - nm) * LOG2E);
          float ps = p0 + p1;
#pragma unroll
          for (int off = 1; off < 16; off <<= 1)
            ps += __shfl_xor(ps, off, 32);
          l_row[v] = l_row[v] * alpha + ps;
          m_row[v] = nm;
#pragma unroll
          for (int n = 0; n < 4; ++n) o[n][v] *= alpha;
          Ps[w][v + 8 * half][l16]      = (bf16_t)p0;   // D-layout -> [row][key]
          Ps[w][v + 8 * half][16 + l16] = (bf16_t)p1;
        }
        DS_FENCE();   // wave-local: no block barrier in divergent region

        // P as A-fragment (16 rows x 32 keys).
        v16bf pf;
#pragma unroll
        for (int g = 0; g < 2; ++g)
#pragma unroll
          for (int e = 0; e < 8; ++e)
            pf[g * 8 + e] = Ps[w][l16][half * 8 + g * 16 + e];

        // ---- O += P V : 4 feature tiles; Vt rows are contiguous in key ----
#pragma unroll
        for (int n = 0; n < 4; ++n) {
          v16bf vb;
          const int fr = n * 16 + l16;        // feature row of Vt
#pragma unroll
          for (int g = 0; g < 2; ++g)
#pragma unroll
            for (int e = 0; e < 8; ++e)
              vb[g * 8 + e] = Vt[cur][fr][s32 * 32 + half * 8 + g * 16 + e];
          o[n] = WMMA_BF16(pf, vb, o[n]);
        }
      }
    }
    __syncthreads();
    vcur[0] = vnxt[0];
    vcur[1] = vnxt[1];
  }

  // Normalize and write bf16 [B, T, C] (t-major for the proj GEMM).
#pragma unroll
  for (int v = 0; v < 8; ++v) {
    const float inv = 1.0f / l_row[v];
    const int t = q0 + v + 8 * half;
    const long base = ((long)b * 2048 + t) * 1024 + h * 64;
#pragma unroll
    for (int n = 0; n < 4; ++n)
      Ob[base + n * 16 + l16] = (bf16_t)(o[n][v] * inv);
  }
}

// ---------------------------------------------------------------------------
// Kernel 3: projection GEMM. A = attn bf16 [8192,1024], B = wpb bf16
// [1024,1024]. Out fp32 [8192,1024]. Same pipeline as kernel 1.
// ---------------------------------------------------------------------------
__global__ __launch_bounds__(256)
void proj_gemm_kernel(const bf16_t* __restrict__ A, const bf16_t* __restrict__ W,
                      float* __restrict__ Out)
{
  __shared__ __align__(16) bf16_t As[2][128][48];
  __shared__ __align__(16) bf16_t Bs[2][128][48];

  const int tid  = threadIdx.x;
  const int lane = tid & 31;
  const int w    = tid >> 5;
  const int wm   = w >> 2, wn = w & 3;
  const int half = lane >> 4, l16 = lane & 15;
  const int mb   = blockIdx.x;   // 0..63
  const int nb   = blockIdx.y;   // 0..7

  v8f acc[4][2];
#pragma unroll
  for (int i = 0; i < 4; ++i)
#pragma unroll
    for (int j = 0; j < 2; ++j)
#pragma unroll
      for (int e = 0; e < 8; ++e) acc[i][j][e] = 0.0f;

  stage_tile_async(A, 1024, (long)mb * 128, 0, As[0], tid);
  stage_tile_async(W, 1024, (long)nb * 128, 0, Bs[0], tid);

  for (int kt = 0; kt < 32; ++kt) {
    const int cur = kt & 1;
    if (kt + 1 < 32) {
      stage_tile_async(A, 1024, (long)mb * 128, (kt + 1) * 32, As[1 - cur], tid);
      stage_tile_async(W, 1024, (long)nb * 128, (kt + 1) * 32, Bs[1 - cur], tid);
      ASYNC_WAIT("0x4");
    } else {
      ASYNC_WAIT("0x0");
    }
    __syncthreads();

    v16bf afr[4], bfr[2];
#pragma unroll
    for (int i = 0; i < 4; ++i) afr[i] = load_frag(As[cur], wm * 64 + i * 16, l16, half);
#pragma unroll
    for (int j = 0; j < 2; ++j) bfr[j] = load_frag(Bs[cur], wn * 32 + j * 16, l16, half);
#pragma unroll
    for (int i = 0; i < 4; ++i)
#pragma unroll
      for (int j = 0; j < 2; ++j)
        acc[i][j] = WMMA_BF16(afr[i], bfr[j], acc[i][j]);
    __syncthreads();
  }

#pragma unroll
  for (int i = 0; i < 4; ++i) {
#pragma unroll
    for (int j = 0; j < 2; ++j) {
      const int n = nb * 128 + wn * 32 + j * 16 + l16;
#pragma unroll
      for (int v = 0; v < 8; ++v) {
        const int m = mb * 128 + wm * 64 + i * 16 + v + 8 * half;
        Out[(long)m * 1024 + n] = acc[i][j][v];
      }
    }
  }
}

// ---------------------------------------------------------------------------
extern "C" void kernel_launch(void* const* d_in, const int* in_sizes, int n_in,
                              void* d_out, int out_size, void* d_ws, size_t ws_size,
                              hipStream_t stream) {
  const float* x      = (const float*)d_in[0];   // [4,2048,1024]
  const float* w_qkv  = (const float*)d_in[1];   // [3072,1024]
  const float* w_proj = (const float*)d_in[2];   // [1024,1024]
  float* out = (float*)d_out;                    // [4,2048,1024] f32

  // Workspace layout (bf16 elems):
  //   xb | wqb | wpb | Q | K | V | Ob
  bf16_t* ws  = (bf16_t*)d_ws;
  bf16_t* xb  = ws;                                 // 8388608
  bf16_t* wqb = xb + QKV_ELEMS;                     // 3145728
  bf16_t* wpb = wqb + 3145728;                      // 1048576
  bf16_t* Qb  = wpb + 1048576;
  bf16_t* Kb  = Qb + QKV_ELEMS;
  bf16_t* Vb  = Kb + QKV_ELEMS;
  bf16_t* Ob  = Vb + QKV_ELEMS;                     // total ~88 MB

  dim3 blk(256);
  cvt_bf16_kernel<<<dim3(4096), blk, 0, stream>>>(x, xb);        // 8.4M elems
  cvt_bf16_kernel<<<dim3(1536), blk, 0, stream>>>(w_qkv, wqb);   // 3.1M elems
  cvt_bf16_kernel<<<dim3(512),  blk, 0, stream>>>(w_proj, wpb);  // 1.0M elems
  qkv_gemm_kernel<<<dim3(64, 24), blk, 0, stream>>>(xb, wqb, Qb, Kb, Vb);
  flash_attn_kernel<<<dim3(1024), blk, 0, stream>>>(Qb, Kb, Vb, Ob);
  proj_gemm_kernel<<<dim3(64, 8), blk, 0, stream>>>(Ob, wpb, out);
}
